// ConsistencyLoss_40063454937742
// MI455X (gfx1250) — matrix operand order repaired and checked by reading, not compiled, not run on hardware
//
#include <hip/hip_runtime.h>
#include <math.h>

// Problem constants (from reference)
#define NB 64
#define NC 256
#define NN 784          // H*W = 28*28
#define NWID 28
#define EPSV 1e-8f

// Tiling
#define BM 64           // rows per block (n index of sim matrix)
#define BN 128          // cols per block (m index)
#define BK 32           // K chunk (matches wmma 16x16x32 bf16)
#define NKCH (NC/BK)    // 8 K chunks
#define PK 40           // padded LDS pitch in bf16 elems (80B, 16B aligned rows)
#define GM 13           // ceil(784/64)
#define GN 7            // ceil(784/128)
#define NTILES (GM*GN)  // 91 tiles per (pair,batch)

// Workspace layout (float offsets). Total ~896 KB.
#define OFF_NORM 0                   // 4 arrays of NB*NN: ny1, nz2, ny2, nz1
#define OFF_BIN  (4*NB*NN)           // bin1[64], bin2[64]
#define OFF_PART (OFF_BIN + 2*NB)    // NTILES*NB*2 slots * 2 floats (sum,cnt)

typedef __attribute__((ext_vector_type(16))) __bf16 v16bf;
typedef __attribute__((ext_vector_type(8)))  float  v8f;

__device__ __forceinline__ v16bf frag_a(const __bf16* base) {
  // elements 0..7 at base, elements 8..15 at base+16 elems (+32B)
  union { uint4 q[2]; v16bf v; } u;
  u.q[0] = *reinterpret_cast<const uint4*>(base);
  u.q[1] = *reinterpret_cast<const uint4*>(base + 16);
  return u.v;
}
__device__ __forceinline__ v16bf frag_b(const __bf16* base) {
  // elements 0..7 at base, elements 8..15 at base+16 elems (+32B)
  union { uint4 q[2]; v16bf v; } u;
  u.q[0] = *reinterpret_cast<const uint4*>(base);
  u.q[1] = *reinterpret_cast<const uint4*>(base + 16);
  return u.v;
}

// ---------------- bin sizes: bin = || grid[:, 1,1] - grid[:, 0,0] || ----------
__global__ void bins_kernel(const float* __restrict__ g1,
                            const float* __restrict__ g2,
                            float* __restrict__ ws) {
  int t = threadIdx.x;              // 0..127
  const float* g = (t < NB) ? g1 : g2;
  int b = t & 63;
  const float* base = g + (size_t)b * 2 * NN; // [2, H, W] for batch b
  float dy = base[NWID + 1] - base[0];        // channel 0 (y)
  float dx = base[NN + NWID + 1] - base[NN];  // channel 1 (x)
  ws[OFF_BIN + t] = sqrtf(dy * dy + dx * dx);
}

// ---------------- per-column L2 norms of all four tensors --------------------
__global__ void norm_kernel(const float* __restrict__ y1,
                            const float* __restrict__ y2,
                            const float* __restrict__ z1,
                            const float* __restrict__ z2,
                            float* __restrict__ ws) {
  int g = blockIdx.x * blockDim.x + threadIdx.x;
  if (g >= NB * NN) return;
  int b = g / NN, n = g % NN;
  size_t base = (size_t)b * NC * NN + n;
  const float* p1 = y1 + base;
  const float* p2 = z2 + base;
  const float* p3 = y2 + base;
  const float* p4 = z1 + base;
  float s1 = 0.f, s2 = 0.f, s3 = 0.f, s4 = 0.f;
  for (int c = 0; c < NC; ++c) {
    float a = p1[(size_t)c * NN]; s1 += a * a;
    float d = p2[(size_t)c * NN]; s2 += d * d;
    float e = p3[(size_t)c * NN]; s3 += e * e;
    float f = p4[(size_t)c * NN]; s4 += f * f;
  }
  ws[OFF_NORM + 0 * NB * NN + g] = sqrtf(s1);  // ||y1||
  ws[OFF_NORM + 1 * NB * NN + g] = sqrtf(s2);  // ||z2||
  ws[OFF_NORM + 2 * NB * NN + g] = sqrtf(s3);  // ||y2||
  ws[OFF_NORM + 3 * NB * NN + g] = sqrtf(s4);  // ||z1||
}

// ------------- fused GEMM (bf16 hi/lo split WMMA) + mask + reduction ---------
// Double-buffered LDS; global loads for chunk k+1 issued before computing on k.
// Staging: each thread owns (1 row x 4 K) groups -> packed 8B LDS stores.
__global__ __launch_bounds__(256)
void gemm_mask_kernel(const float* __restrict__ y1, const float* __restrict__ y2,
                      const float* __restrict__ z1, const float* __restrict__ z2,
                      const float* __restrict__ g1, const float* __restrict__ g2,
                      float* __restrict__ ws) {
  __shared__ __bf16 sA[2][2][BM * PK];   // [buf][hi/lo]
  __shared__ __bf16 sB[2][2][BN * PK];
  __shared__ float red[512];

  const int tid  = threadIdx.x;
  const int lane = tid & 31;
  const int wave = tid >> 5;
  const int z = blockIdx.z;      // 0..127
  const int p = z >> 6;          // pair: 0 -> (y1,z2,bin1), 1 -> (y2,z1,bin2)
  const int b = z & 63;
  const int rowBlk = blockIdx.x * BM;   // n base
  const int colBlk = blockIdx.y * BN;   // m base

  const float* Ap = (p ? y2 : y1) + (size_t)b * NC * NN;
  const float* Bp = (p ? z1 : z2) + (size_t)b * NC * NN;
  const float* nA = ws + OFF_NORM + (p ? 2 : 0) * NB * NN + (size_t)b * NN;
  const float* nBm = ws + OFF_NORM + (p ? 3 : 1) * NB * NN + (size_t)b * NN;
  const float* G1 = g1 + (size_t)b * 2 * NN;  // [0..NN)=y, [NN..2NN)=x
  const float* G2 = g2 + (size_t)b * 2 * NN;
  const float bin = ws[OFF_BIN + p * NB + b];
  const float thr2 = 0.49f * bin * bin;       // (0.7*bin)^2

  const int wm = wave >> 2;  // 0..1 : 32-row slab
  const int wn = wave & 3;   // 0..3 : 32-col slab

  // Per-thread staging coordinates: groups of (1 row x 4 consecutive K).
  // A: 64 rows x 32 K = 2048 elems -> 2 groups/thread.
  int aNl[2], aC4[2];
  bool aOk[2];
#pragma unroll
  for (int i = 0; i < 2; ++i) {
    int v = tid + 256 * i;            // 0..511
    aNl[i] = v & 63;                  // local n
    aC4[i] = (v >> 6) * 4;            // K group base: 0,4,...,28
    aOk[i] = (rowBlk + aNl[i]) < NN;
  }
  // B: 128 rows x 32 K = 4096 elems -> 4 groups/thread.
  int bMl[4], bC4[4];
  bool bOk[4];
#pragma unroll
  for (int i = 0; i < 4; ++i) {
    int v = tid + 256 * i;            // 0..1023
    bMl[i] = v & 127;                 // local m
    bC4[i] = (v >> 7) * 4;            // K group base
    bOk[i] = (colBlk + bMl[i]) < NN;
  }

  float av[2][4], bv[4][4];

  // ---- chunk loader: per-lane b32 loads, coalesced over rows ----
  auto load_chunk = [&](int kc) {
#pragma unroll
    for (int i = 0; i < 2; ++i) {
      const float* src = Ap + (size_t)(kc + aC4[i]) * NN + rowBlk + aNl[i];
#pragma unroll
      for (int j = 0; j < 4; ++j)
        av[i][j] = aOk[i] ? src[(size_t)j * NN] : 0.0f;
    }
#pragma unroll
    for (int i = 0; i < 4; ++i) {
      const float* src = Bp + (size_t)(kc + bC4[i]) * NN + colBlk + bMl[i];
#pragma unroll
      for (int j = 0; j < 4; ++j)
        bv[i][j] = bOk[i] ? src[(size_t)j * NN] : 0.0f;
    }
  };
  // ---- convert (bf16 hi/lo split) + packed 8B LDS stores ----
  auto stage_chunk = [&](int buf) {
#pragma unroll
    for (int i = 0; i < 2; ++i) {
      union { __bf16 h[4]; uint2 q; } uh, ul;
#pragma unroll
      for (int j = 0; j < 4; ++j) {
        float f = av[i][j];
        __bf16 h = (__bf16)f;
        uh.h[j] = h;
        ul.h[j] = (__bf16)(f - (float)h);
      }
      int off = aNl[i] * PK + aC4[i];        // 8B aligned
      *reinterpret_cast<uint2*>(&sA[buf][0][off]) = uh.q;
      *reinterpret_cast<uint2*>(&sA[buf][1][off]) = ul.q;
    }
#pragma unroll
    for (int i = 0; i < 4; ++i) {
      union { __bf16 h[4]; uint2 q; } uh, ul;
#pragma unroll
      for (int j = 0; j < 4; ++j) {
        float f = bv[i][j];
        __bf16 h = (__bf16)f;
        uh.h[j] = h;
        ul.h[j] = (__bf16)(f - (float)h);
      }
      int off = bMl[i] * PK + bC4[i];
      *reinterpret_cast<uint2*>(&sB[buf][0][off]) = uh.q;
      *reinterpret_cast<uint2*>(&sB[buf][1][off]) = ul.q;
    }
  };

  v8f acc[2][2];
  const v8f vzero = {0.f,0.f,0.f,0.f,0.f,0.f,0.f,0.f};
  acc[0][0] = vzero; acc[0][1] = vzero; acc[1][0] = vzero; acc[1][1] = vzero;

  const int row16 = lane & 15;
  const int koA   = (lane < 16) ? 0 : 8;    // A: interleaved K halves per lane group
  const int koB   = (lane < 16) ? 0 : 8;    // B: interleaved K halves per lane group

  // Prologue: stage chunk 0.
  load_chunk(0);
  stage_chunk(0);
  __syncthreads();

#pragma unroll
  for (int kk = 0; kk < NKCH; ++kk) {
    const int cur = kk & 1;
    const bool more = (kk + 1) < NKCH;
    if (more) load_chunk((kk + 1) * BK);   // overlap VMEM with WMMA below

    v16bf aH[2], aL[2], bH[2], bL[2];
#pragma unroll
    for (int i = 0; i < 2; ++i) {
      int r = (wm * 32 + 16 * i + row16) * PK + koA;
      aH[i] = frag_a(&sA[cur][0][r]);
      aL[i] = frag_a(&sA[cur][1][r]);
    }
#pragma unroll
    for (int j = 0; j < 2; ++j) {
      int r = (wn * 32 + 16 * j + row16) * PK + koB;
      bH[j] = frag_b(&sB[cur][0][r]);
      bL[j] = frag_b(&sB[cur][1][r]);
    }

    // fp32-accurate dot via bf16 split: hi*hi + hi*lo + lo*hi
#pragma unroll
    for (int i = 0; i < 2; ++i)
#pragma unroll
      for (int j = 0; j < 2; ++j) {
        acc[i][j] = __builtin_amdgcn_wmma_f32_16x16x32_bf16(
            false, aH[i], false, bH[j], (short)0, acc[i][j], false, false);
        acc[i][j] = __builtin_amdgcn_wmma_f32_16x16x32_bf16(
            false, aH[i], false, bL[j], (short)0, acc[i][j], false, false);
        acc[i][j] = __builtin_amdgcn_wmma_f32_16x16x32_bf16(
            false, aL[i], false, bH[j], (short)0, acc[i][j], false, false);
      }

    if (more) {
      __syncthreads();           // all waves done reading; safe to fill other buf
      stage_chunk(cur ^ 1);
      __syncthreads();           // publish next buf
    }
  }

  // Epilogue: cosine + distance mask, accumulated in registers.
  // C/D layout: VGPR r -> M = r (+8 for lanes 16..31), N = lane & 15.
  float lsum = 0.f, lcnt = 0.f;
  const int rowOff = (lane >= 16) ? 8 : 0;
#pragma unroll
  for (int j = 0; j < 2; ++j) {
    int mIdx = colBlk + wn * 32 + 16 * j + row16;
    if (mIdx < NN) {
      float nb_ = nBm[mIdx];
      float g2y = G2[mIdx];
      float g2x = G2[NN + mIdx];
#pragma unroll
      for (int i = 0; i < 2; ++i) {
        int nBase = rowBlk + wm * 32 + 16 * i + rowOff;
#pragma unroll
        for (int r = 0; r < 8; ++r) {
          int nIdx = nBase + r;
          if (nIdx < NN) {
            float dy = G1[nIdx] - g2y;
            float dx = G1[NN + nIdx] - g2x;
            if (dy * dy + dx * dx <= thr2) {
              float den = fmaxf(nA[nIdx] * nb_, EPSV);
              lsum += acc[i][j][r] / den;
              lcnt += 1.f;
            }
          }
        }
      }
    }
  }

  // Deterministic block tree reduction, one partial slot per block.
  red[tid] = lsum;
  red[256 + tid] = lcnt;
  __syncthreads();
  for (int s = 128; s > 0; s >>= 1) {
    if (tid < s) {
      red[tid] += red[tid + s];
      red[256 + tid] += red[256 + tid + s];
    }
    __syncthreads();
  }
  if (tid == 0) {
    int slot = (blockIdx.z * NTILES + blockIdx.y * GM + blockIdx.x) * 2;
    ws[OFF_PART + slot]     = red[0];
    ws[OFF_PART + slot + 1] = red[256];
  }
}

// ---------------- deterministic final reduction -> scalar loss ---------------
__global__ void reduce_kernel(const float* __restrict__ ws, float* __restrict__ out) {
  __shared__ float s[1024];
  int t = threadIdx.x;
  const int half = NTILES * NB;   // slots per pair
  float s0 = 0.f, c0 = 0.f, s1 = 0.f, c1 = 0.f;
  for (int i = t; i < half; i += 256) {
    s0 += ws[OFF_PART + i * 2];
    c0 += ws[OFF_PART + i * 2 + 1];
    s1 += ws[OFF_PART + (half + i) * 2];
    c1 += ws[OFF_PART + (half + i) * 2 + 1];
  }
  s[t] = s0; s[256 + t] = c0; s[512 + t] = s1; s[768 + t] = c1;
  __syncthreads();
  for (int st = 128; st > 0; st >>= 1) {
    if (t < st) {
      s[t] += s[t + st];
      s[256 + t] += s[256 + t + st];
      s[512 + t] += s[512 + t + st];
      s[768 + t] += s[768 + t + st];
    }
    __syncthreads();
  }
  if (t == 0) {
    float m1 = s[0]   / fmaxf(s[256], 1.0f);
    float m2 = s[512] / fmaxf(s[768], 1.0f);
    out[0] = -(m1 + m2);
  }
}

extern "C" void kernel_launch(void* const* d_in, const int* in_sizes, int n_in,
                              void* d_out, int out_size, void* d_ws, size_t ws_size,
                              hipStream_t stream) {
  (void)in_sizes; (void)n_in; (void)out_size; (void)ws_size;
  const float* y1 = (const float*)d_in[0];
  const float* y2 = (const float*)d_in[1];
  const float* z1 = (const float*)d_in[2];
  const float* z2 = (const float*)d_in[3];
  const float* g1 = (const float*)d_in[4];
  const float* g2 = (const float*)d_in[5];
  float* ws  = (float*)d_ws;   // needs ~896 KB
  float* out = (float*)d_out;

  bins_kernel<<<1, 128, 0, stream>>>(g1, g2, ws);
  norm_kernel<<<(NB * NN + 255) / 256, 256, 0, stream>>>(y1, y2, z1, z2, ws);
  dim3 grid(GM, GN, 2 * NB);
  gemm_mask_kernel<<<grid, 256, 0, stream>>>(y1, y2, z1, z2, g1, g2, ws);
  reduce_kernel<<<1, 256, 0, stream>>>(ws, out);
}
